// Classifier_45681272160376
// MI455X (gfx1250) — compile-verified, matrix operand order checked by reading, not run on hardware
//
#include <hip/hip_runtime.h>
#include <stdint.h>

#define B_   16
#define C_   512
#define U_   256
#define P_   1024
#define OUTD 10

#define BM 128
#define BN 128
#define BK 32
#define LDT 40   // LDS row stride in ushorts (mult of 8 -> 16B aligned rows)

typedef __attribute__((ext_vector_type(16))) __bf16 v16bf;
typedef __attribute__((ext_vector_type(8)))  float  v8f;
typedef unsigned int u32x4 __attribute__((ext_vector_type(4)));
typedef int          i32x4 __attribute__((ext_vector_type(4)));
typedef int          i32x8 __attribute__((ext_vector_type(8)));

union FragAB { v16bf v; uint4 q[2]; };
union FragC  { v8f   v; float  f[8]; };

__device__ __forceinline__ unsigned short f2bf(float f) {
  union { float f; unsigned int u; } x; x.f = f;
  unsigned int r = 0x7FFFu + ((x.u >> 16) & 1u);   // round-to-nearest-even
  return (unsigned short)((x.u + r) >> 16);
}

__device__ __forceinline__ v8f wmma_bf16(v16bf a, v16bf b, v8f c) {
  return __builtin_amdgcn_wmma_f32_16x16x32_bf16(false, a, false, b, (short)0, c,
                                                 false, false);
}

// Load a 16x32 bf16 fragment from an LDS tile stored row-major (row, k), k contiguous.
__device__ __forceinline__ v16bf ld_frag(const unsigned short* base, int row, int ld) {
  int lane = threadIdx.x & 31;
  int r = row + (lane & 15);
  int h = lane >> 4;
  FragAB f;
  f.q[0] = *(const uint4*)(base + r * ld + h * 8);
  f.q[1] = *(const uint4*)(base + r * ld + 16 + h * 8);
  return f.v;
}

// ---------------------------------------------------------------------------
// Tensor Data Mover: async-load a 2D bf16 tile (tile_d0 contiguous elems x
// tile_d1 rows, global row stride = stride_elems) into LDS at lds_off,
// with LDS row padding to reproduce the LDT=40 (80-byte) padded layout:
// pad_interval code 3 = every 16 DWORDs (64B row), pad_amount code 3 = 4 DWORDs.
// D# bit layout per CDNA5 ISA.
// ---------------------------------------------------------------------------
__device__ __forceinline__ void tdm_load_2d_bf16(
    unsigned lds_off, const void* gptr,
    unsigned tile_d0, unsigned tile_d1, unsigned stride_elems) {
  unsigned long long ga = (unsigned long long)(uintptr_t)gptr;
  u32x4 g0;
  g0[0] = 1u;                                       // count=1, user descriptor
  g0[1] = lds_off;                                  // lds_addr (bytes)
  g0[2] = (unsigned)(ga & 0xFFFFFFFFu);             // global_addr[31:0]
  g0[3] = (unsigned)((ga >> 32) & 0x01FFFFFFu)      // global_addr[56:32]
        | (2u << 30);                               // type=2 ("image")
  i32x8 g1;
  g1[0] = (int)((1u << 16)                          // data_size = 1 -> 2 bytes
        | (1u << 20)                                // pad_enable
        | (3u << 22)                                // pad_interval: 16 DWORDs
        | (3u << 25));                              // pad_amount: 4 DWORDs
  g1[1] = (int)((tile_d0 & 0xFFFFu) << 16);         // tensor_dim0[15:0]  (bits 63:48)
  g1[2] = (int)(((tile_d0 >> 16) & 0xFFFFu)         // tensor_dim0[31:16]
        | ((tile_d1 & 0xFFFFu) << 16));             // tensor_dim1[15:0]  (bits 95:80)
  g1[3] = (int)(((tile_d1 >> 16) & 0xFFFFu)         // tensor_dim1[31:16]
        | ((tile_d0 & 0xFFFFu) << 16));             // tile_dim0          (bits 127:112)
  g1[4] = (int)(tile_d1 & 0xFFFFu);                 // tile_dim1; tile_dim2 = 0
  g1[5] = (int)stride_elems;                        // tensor_dim0_stride[31:0]
  g1[6] = 0;                                        // stride0[47:32], dim1_stride[15:0]
  g1[7] = 0;
  i32x4 gz = {0, 0, 0, 0};
#if defined(__clang_major__) && (__clang_major__ >= 23)
  i32x8 gz8 = {0, 0, 0, 0, 0, 0, 0, 0};
  __builtin_amdgcn_tensor_load_to_lds(g0, g1, gz, gz, gz8, 0);
#else
  __builtin_amdgcn_tensor_load_to_lds(g0, g1, gz, gz, 0);
#endif
}

__device__ __forceinline__ float block_max(float v, float* red) {
  int t = threadIdx.x;
  red[t] = v; __syncthreads();
  for (int o = 128; o; o >>= 1) { if (t < o) red[t] = fmaxf(red[t], red[t + o]); __syncthreads(); }
  float r = red[0]; __syncthreads(); return r;
}
__device__ __forceinline__ float block_sum(float v, float* red) {
  int t = threadIdx.x;
  red[t] = v; __syncthreads();
  for (int o = 128; o; o >>= 1) { if (t < o) red[t] += red[t + o]; __syncthreads(); }
  float r = red[0]; __syncthreads(); return r;
}

// ---------------------------------------------------------------------------
// K1: four conv1x1 GEMMs.  Out[b](256,1024) = W(256,512) @ X[b](512,1024) + bias
// grid: (8 N-tiles, 2 M-tiles, 64 = b*4+which), block 256
// ---------------------------------------------------------------------------
__global__ __launch_bounds__(256) void k_conv4(
    const float* xs, const float* ys, const float* xe, const float* ye,
    const float* w1, const float* b1, const float* w2, const float* b2,
    const float* w3, const float* b3, const float* w4, const float* b4,
    float* outF, unsigned short* nat, unsigned short* tr) {
  __shared__ __align__(16) unsigned short As[BM * LDT];
  __shared__ __align__(16) unsigned short Bs[BN * LDT];

  const size_t NE = (size_t)B_ * U_ * P_;
  const int which = blockIdx.z & 3;
  const int b = blockIdx.z >> 2;
  const float *X, *W, *BI;
  if (which == 0)      { X = xs; W = w1; BI = b1; }
  else if (which == 1) { X = ys; W = w2; BI = b2; }
  else if (which == 2) { X = xe; W = w3; BI = b3; }
  else                 { X = ye; W = w4; BI = b4; }
  float* dst = outF + 160 + NE + (size_t)which * NE;   // xs_c, ys_c, xe_c, ye_c
  unsigned short* natW = nat + (size_t)which * NE;
  unsigned short* trW  = tr  + (size_t)which * NE;
  const float* Xb = X + (size_t)b * C_ * P_;

  const int m0 = blockIdx.y * BM;
  const int n0 = blockIdx.x * BN;
  const int tid = threadIdx.x;
  const int wid = tid >> 5;
  const int wm0 = (wid & 3) * 32;
  const int wn0 = (wid >> 2) * 64;

  v8f zero = {};
  v8f acc[2][4];
  for (int i = 0; i < 2; i++) for (int j = 0; j < 4; j++) acc[i][j] = zero;

  const int arow = tid >> 1, acol = (tid & 1) * 16;     // A: 128 rows x 32 k
  const int bkr = tid >> 3,  bc0  = (tid & 7) * 16;     // B: 32 k x 128 n

  for (int kk = 0; kk < C_; kk += BK) {
    __syncthreads();
    { // A tile: weights (row-major, K contiguous) -> bf16 LDS
      const float4* s4 = (const float4*)(W + (size_t)(m0 + arow) * C_ + kk + acol);
      unsigned short* d = As + arow * LDT + acol;
      #pragma unroll
      for (int i = 0; i < 4; i++) {
        float4 v = s4[i];
        d[i * 4 + 0] = f2bf(v.x); d[i * 4 + 1] = f2bf(v.y);
        d[i * 4 + 2] = f2bf(v.z); d[i * 4 + 3] = f2bf(v.w);
      }
    }
    { // B tile: X (C,P) -> transpose to (n,k) bf16 LDS
      const float4* s4 = (const float4*)(Xb + (size_t)(kk + bkr) * P_ + n0 + bc0);
      #pragma unroll
      for (int i = 0; i < 4; i++) {
        float4 v = s4[i];
        Bs[(bc0 + i * 4 + 0) * LDT + bkr] = f2bf(v.x);
        Bs[(bc0 + i * 4 + 1) * LDT + bkr] = f2bf(v.y);
        Bs[(bc0 + i * 4 + 2) * LDT + bkr] = f2bf(v.z);
        Bs[(bc0 + i * 4 + 3) * LDT + bkr] = f2bf(v.w);
      }
      if (kk + BK < C_)
        __builtin_prefetch(Xb + (size_t)(kk + BK + bkr) * P_ + n0 + bc0, 0, 1);
    }
    __syncthreads();
    v16bf a0 = ld_frag(As, wm0, LDT);
    v16bf a1 = ld_frag(As, wm0 + 16, LDT);
    v16bf bb[4];
    #pragma unroll
    for (int j = 0; j < 4; j++) bb[j] = ld_frag(Bs, wn0 + j * 16, LDT);
    #pragma unroll
    for (int j = 0; j < 4; j++) {
      acc[0][j] = wmma_bf16(a0, bb[j], acc[0][j]);
      acc[1][j] = wmma_bf16(a1, bb[j], acc[1][j]);
    }
  }

  const int lane = tid & 31, ln = lane & 15, lh = lane >> 4;
  #pragma unroll
  for (int mi = 0; mi < 2; mi++)
    #pragma unroll
    for (int ni = 0; ni < 4; ni++) {
      FragC c; c.v = acc[mi][ni];
      int p = n0 + wn0 + ni * 16 + ln;
      #pragma unroll
      for (int r = 0; r < 8; r++) {
        int u = m0 + wm0 + mi * 16 + r + 8 * lh;
        float val = c.f[r] + BI[u];
        size_t idx = ((size_t)b * U_ + u) * P_ + p;
        dst[idx] = val;
        unsigned short bf = f2bf(val);
        natW[idx] = bf;
        trW[((size_t)b * P_ + p) * U_ + u] = bf;
      }
    }
}

// ---------------------------------------------------------------------------
// K2: affinity scores S[b](1024,1024) = A(P,U) @ B(P,U)^T  (both K=U contiguous)
// A/B tiles are pure bf16 copies -> fetched by the Tensor Data Mover.
// grid: (8,8,32 = b*2+sel)
// ---------------------------------------------------------------------------
__global__ __launch_bounds__(256) void k_affinity(
    const unsigned short* tr, float* Sx, float* Sy) {
  __shared__ __align__(16) unsigned short As[BM * LDT];
  __shared__ __align__(16) unsigned short Bs[BN * LDT];
  const size_t NE = (size_t)B_ * U_ * P_;
  const int sel = blockIdx.z & 1;
  const int b = blockIdx.z >> 1;
  const unsigned short* Asrc = tr + (size_t)(sel ? 1 : 0) * NE + (size_t)b * P_ * U_;
  const unsigned short* Bsrc = tr + (size_t)(sel ? 3 : 2) * NE + (size_t)b * P_ * U_;
  float* S = (sel ? Sy : Sx) + (size_t)b * P_ * P_;

  const int m0 = blockIdx.y * BM, n0 = blockIdx.x * BN;
  const int tid = threadIdx.x, wid = tid >> 5;
  const int wm0 = (wid & 3) * 32, wn0 = (wid >> 2) * 64;
  const unsigned AsOff = (unsigned)(uintptr_t)(As);
  const unsigned BsOff = (unsigned)(uintptr_t)(Bs);

  v8f zero = {};
  v8f acc[2][4];
  for (int i = 0; i < 2; i++) for (int j = 0; j < 4; j++) acc[i][j] = zero;

  for (int kk = 0; kk < U_; kk += BK) {
    __syncthreads();
    if (wid == 0) {                       // one wave drives the TDM
      tdm_load_2d_bf16(AsOff, Asrc + (size_t)m0 * U_ + kk, BK, BM, U_);
      tdm_load_2d_bf16(BsOff, Bsrc + (size_t)n0 * U_ + kk, BK, BN, U_);
      __builtin_amdgcn_s_wait_tensorcnt(0);
    }
    __syncthreads();
    v16bf a0 = ld_frag(As, wm0, LDT);
    v16bf a1 = ld_frag(As, wm0 + 16, LDT);
    v16bf bb[4];
    #pragma unroll
    for (int j = 0; j < 4; j++) bb[j] = ld_frag(Bs, wn0 + j * 16, LDT);
    #pragma unroll
    for (int j = 0; j < 4; j++) {
      acc[0][j] = wmma_bf16(a0, bb[j], acc[0][j]);
      acc[1][j] = wmma_bf16(a1, bb[j], acc[1][j]);
    }
  }

  const int lane = tid & 31, ln = lane & 15, lh = lane >> 4;
  #pragma unroll
  for (int mi = 0; mi < 2; mi++)
    #pragma unroll
    for (int ni = 0; ni < 4; ni++) {
      FragC c; c.v = acc[mi][ni];
      int q = n0 + wn0 + ni * 16 + ln;
      #pragma unroll
      for (int r = 0; r < 8; r++) {
        int p = m0 + wm0 + mi * 16 + r + 8 * lh;
        S[(size_t)p * P_ + q] = c.f[r];
      }
    }
}

// ---------------------------------------------------------------------------
// K3: weight_com row = softmax(softmax(sx) * softmax(sy)); overwrites Sx.
// grid: 16384 rows, block 256 (4 elems/thread)
// ---------------------------------------------------------------------------
__global__ __launch_bounds__(256) void k_softmax(float* Sx, const float* Sy) {
  __shared__ float red[256];
  const size_t row = blockIdx.x;
  float* px = Sx + row * P_;
  const float* py = Sy + row * P_;
  const int t = threadIdx.x;
  float4 vx = ((const float4*)px)[t];
  float4 vy = ((const float4*)py)[t];

  float mx = block_max(fmaxf(fmaxf(vx.x, vx.y), fmaxf(vx.z, vx.w)), red);
  float ex0 = __expf(vx.x - mx), ex1 = __expf(vx.y - mx);
  float ex2 = __expf(vx.z - mx), ex3 = __expf(vx.w - mx);
  float sx = block_sum(ex0 + ex1 + ex2 + ex3, red);

  float my = block_max(fmaxf(fmaxf(vy.x, vy.y), fmaxf(vy.z, vy.w)), red);
  float ey0 = __expf(vy.x - my), ey1 = __expf(vy.y - my);
  float ey2 = __expf(vy.z - my), ey3 = __expf(vy.w - my);
  float sy = block_sum(ey0 + ey1 + ey2 + ey3, red);

  float inv = 1.0f / (sx * sy);
  float z0 = ex0 * ey0 * inv, z1 = ex1 * ey1 * inv;
  float z2 = ex2 * ey2 * inv, z3 = ex3 * ey3 * inv;

  float mz = block_max(fmaxf(fmaxf(z0, z1), fmaxf(z2, z3)), red);
  float e0 = __expf(z0 - mz), e1 = __expf(z1 - mz);
  float e2 = __expf(z2 - mz), e3 = __expf(z3 - mz);
  float sz = block_sum(e0 + e1 + e2 + e3, red);
  float r = 1.0f / sz;
  float4 w; w.x = e0 * r; w.y = e1 * r; w.z = e2 * r; w.w = e3 * r;
  ((float4*)px)[t] = w;
}

// ---------------------------------------------------------------------------
// K4: x_dif conv.  Out[b](256,1024) = w5(256,1024) @ concat[xs,xe,ys,ye](1024,1024)
// grid: (8,2,16 = b)
// ---------------------------------------------------------------------------
__global__ __launch_bounds__(256) void k_xdif(
    const float* w5, const float* b5, const unsigned short* nat,
    float* xdifF, unsigned short* xdifN) {
  __shared__ __align__(16) unsigned short As[BM * LDT];
  __shared__ __align__(16) unsigned short Bs[BN * LDT];
  const size_t NE = (size_t)B_ * U_ * P_;
  const int b = blockIdx.z;
  const unsigned short* srcs[4] = { nat, nat + 2 * NE, nat + 1 * NE, nat + 3 * NE };

  const int m0 = blockIdx.y * BM, n0 = blockIdx.x * BN;
  const int tid = threadIdx.x, wid = tid >> 5;
  const int wm0 = (wid & 3) * 32, wn0 = (wid >> 2) * 64;
  const int arow = tid >> 1, acol = (tid & 1) * 16;
  const int bkr = tid >> 3,  bc0  = (tid & 7) * 16;

  v8f zero = {};
  v8f acc[2][4];
  for (int i = 0; i < 2; i++) for (int j = 0; j < 4; j++) acc[i][j] = zero;

  for (int kk = 0; kk < 4 * U_; kk += BK) {
    __syncthreads();
    { // A: w5 row-major fp32 -> bf16
      const float4* s4 = (const float4*)(w5 + (size_t)(m0 + arow) * (4 * U_) + kk + acol);
      unsigned short* d = As + arow * LDT + acol;
      #pragma unroll
      for (int i = 0; i < 4; i++) {
        float4 v = s4[i];
        d[i * 4 + 0] = f2bf(v.x); d[i * 4 + 1] = f2bf(v.y);
        d[i * 4 + 2] = f2bf(v.z); d[i * 4 + 3] = f2bf(v.w);
      }
    }
    { // B: gather from the 4 bf16 conv outputs, transpose into (n,k)
      int kglob = kk + bkr;
      const unsigned short* sp =
          srcs[kglob >> 8] + ((size_t)b * U_ + (kglob & 255)) * P_ + n0 + bc0;
      union { uint4 q[2]; unsigned short s[16]; } tb;
      tb.q[0] = ((const uint4*)sp)[0];
      tb.q[1] = ((const uint4*)sp)[1];
      #pragma unroll
      for (int i = 0; i < 16; i++) Bs[(bc0 + i) * LDT + bkr] = tb.s[i];
    }
    __syncthreads();
    v16bf a0 = ld_frag(As, wm0, LDT);
    v16bf a1 = ld_frag(As, wm0 + 16, LDT);
    v16bf bb[4];
    #pragma unroll
    for (int j = 0; j < 4; j++) bb[j] = ld_frag(Bs, wn0 + j * 16, LDT);
    #pragma unroll
    for (int j = 0; j < 4; j++) {
      acc[0][j] = wmma_bf16(a0, bb[j], acc[0][j]);
      acc[1][j] = wmma_bf16(a1, bb[j], acc[1][j]);
    }
  }

  const int lane = tid & 31, ln = lane & 15, lh = lane >> 4;
  #pragma unroll
  for (int mi = 0; mi < 2; mi++)
    #pragma unroll
    for (int ni = 0; ni < 4; ni++) {
      FragC c; c.v = acc[mi][ni];
      int p = n0 + wn0 + ni * 16 + ln;
      #pragma unroll
      for (int r = 0; r < 8; r++) {
        int u = m0 + wm0 + mi * 16 + r + 8 * lh;
        float val = c.f[r] + b5[u];
        size_t idx = ((size_t)b * U_ + u) * P_ + p;
        xdifF[idx] = val;
        xdifN[idx] = f2bf(val);
      }
    }
}

// ---------------------------------------------------------------------------
// K5: x_refine + x_final.  Out[b](256,1024) = xdif(256,1024) @ Wcom(1024,1024) + xdif
// A tile is a pure bf16 copy -> Tensor Data Mover; B is fp32->bf16 transpose.
// grid: (8,2,16 = b)
// ---------------------------------------------------------------------------
__global__ __launch_bounds__(256) void k_refine(
    const unsigned short* xdifN, const float* Wcom, const float* xdifF, float* outF) {
  __shared__ __align__(16) unsigned short As[BM * LDT];
  __shared__ __align__(16) unsigned short Bs[BN * LDT];
  const int b = blockIdx.z;
  const unsigned short* Ab = xdifN + (size_t)b * U_ * P_;
  const float* Wb = Wcom + (size_t)b * P_ * P_;

  const int m0 = blockIdx.y * BM, n0 = blockIdx.x * BN;
  const int tid = threadIdx.x, wid = tid >> 5;
  const int wm0 = (wid & 3) * 32, wn0 = (wid >> 2) * 64;
  const int bkr = tid >> 3,  bc0  = (tid & 7) * 16;
  const unsigned AsOff = (unsigned)(uintptr_t)(As);

  v8f zero = {};
  v8f acc[2][4];
  for (int i = 0; i < 2; i++) for (int j = 0; j < 4; j++) acc[i][j] = zero;

  for (int kk = 0; kk < P_; kk += BK) {
    __syncthreads();
    if (wid == 0) {                       // A tile via TDM
      tdm_load_2d_bf16(AsOff, Ab + (size_t)(m0)*P_ + kk, BK, BM, P_);
    }
    { // B: Wcom fp32 (p,q) -> transpose-convert to (n=q, k=p)
      const float4* s4 = (const float4*)(Wb + (size_t)(kk + bkr) * P_ + n0 + bc0);
      #pragma unroll
      for (int i = 0; i < 4; i++) {
        float4 v = s4[i];
        Bs[(bc0 + i * 4 + 0) * LDT + bkr] = f2bf(v.x);
        Bs[(bc0 + i * 4 + 1) * LDT + bkr] = f2bf(v.y);
        Bs[(bc0 + i * 4 + 2) * LDT + bkr] = f2bf(v.z);
        Bs[(bc0 + i * 4 + 3) * LDT + bkr] = f2bf(v.w);
      }
      if (kk + BK < P_)
        __builtin_prefetch(Wb + (size_t)(kk + BK + bkr) * P_ + n0 + bc0, 0, 1);
    }
    if (wid == 0) __builtin_amdgcn_s_wait_tensorcnt(0);
    __syncthreads();
    v16bf a0 = ld_frag(As, wm0, LDT);
    v16bf a1 = ld_frag(As, wm0 + 16, LDT);
    v16bf bb[4];
    #pragma unroll
    for (int j = 0; j < 4; j++) bb[j] = ld_frag(Bs, wn0 + j * 16, LDT);
    #pragma unroll
    for (int j = 0; j < 4; j++) {
      acc[0][j] = wmma_bf16(a0, bb[j], acc[0][j]);
      acc[1][j] = wmma_bf16(a1, bb[j], acc[1][j]);
    }
  }

  const int lane = tid & 31, ln = lane & 15, lh = lane >> 4;
  #pragma unroll
  for (int mi = 0; mi < 2; mi++)
    #pragma unroll
    for (int ni = 0; ni < 4; ni++) {
      FragC c; c.v = acc[mi][ni];
      int p = n0 + wn0 + ni * 16 + ln;
      #pragma unroll
      for (int r = 0; r < 8; r++) {
        int u = m0 + wm0 + mi * 16 + r + 8 * lh;
        size_t idx = ((size_t)b * U_ + u) * P_ + p;
        outF[160 + idx] = c.f[r] + xdifF[idx];   // x_final
      }
    }
}

// ---------------------------------------------------------------------------
// K6: adaptive avg pool (mean over P).  grid: 4096 = b*U+u, block 256
// ---------------------------------------------------------------------------
__global__ __launch_bounds__(256) void k_pool(const float* outF, float* pooled) {
  __shared__ float red[256];
  const size_t row = blockIdx.x;
  const float* src = outF + 160 + row * P_;
  float s = 0.0f;
  for (int i = threadIdx.x; i < P_; i += 256) s += src[i];
  float tot = block_sum(s, red);
  if (threadIdx.x == 0) pooled[row] = tot * (1.0f / P_);
}

// ---------------------------------------------------------------------------
// K7: MLP head.  grid: 16 (b), block 128
// ---------------------------------------------------------------------------
__global__ __launch_bounds__(128) void k_mlp(
    const float* pooled, const float* lw1, const float* lb1,
    const float* lw2, const float* lb2, float* outF) {
  __shared__ float pv[U_];
  __shared__ float hv[128];
  const int b = blockIdx.x;
  for (int i = threadIdx.x; i < U_; i += 128) pv[i] = pooled[b * U_ + i];
  __syncthreads();
  const int j = threadIdx.x;
  float s = lb1[j];
  for (int k = 0; k < U_; k++) s += pv[k] * lw1[j * U_ + k];
  hv[j] = fmaxf(s, 0.0f);
  __syncthreads();
  if (j < OUTD) {
    float o = lb2[j];
    for (int k = 0; k < 128; k++) o += hv[k] * lw2[j * 128 + k];
    outF[b * OUTD + j] = o;
  }
}

// ---------------------------------------------------------------------------
extern "C" void kernel_launch(void* const* d_in, const int* in_sizes, int n_in,
                              void* d_out, int out_size, void* d_ws, size_t ws_size,
                              hipStream_t stream) {
  (void)in_sizes; (void)n_in; (void)out_size; (void)ws_size;
  const float* xs  = (const float*)d_in[0];
  const float* ys  = (const float*)d_in[1];
  const float* xe  = (const float*)d_in[2];
  const float* ye  = (const float*)d_in[3];
  const float* w1  = (const float*)d_in[4];
  const float* b1  = (const float*)d_in[5];
  const float* w2  = (const float*)d_in[6];
  const float* b2  = (const float*)d_in[7];
  const float* w3  = (const float*)d_in[8];
  const float* b3  = (const float*)d_in[9];
  const float* w4  = (const float*)d_in[10];
  const float* b4  = (const float*)d_in[11];
  const float* w5  = (const float*)d_in[12];
  const float* b5  = (const float*)d_in[13];
  const float* lw1 = (const float*)d_in[14];
  const float* lb1 = (const float*)d_in[15];
  const float* lw2 = (const float*)d_in[16];
  const float* lb2 = (const float*)d_in[17];
  float* outF = (float*)d_out;

  const size_t NE = (size_t)B_ * U_ * P_;      // 4,194,304
  const size_t SP = (size_t)B_ * P_ * P_;      // 16,777,216
  char* ws = (char*)d_ws;
  unsigned short* nat   = (unsigned short*)ws;          // 4*NE bf16
  unsigned short* tr    = nat + 4 * NE;                 // 4*NE bf16
  unsigned short* xdifN = tr + 4 * NE;                  // NE bf16
  float* Sx    = (float*)(ws + 9 * NE * sizeof(unsigned short)); // SP (also Wcom)
  float* Sy    = Sx + SP;                               // SP
  float* xdifF = Sy + SP;                               // NE
  float* pooled = xdifF + NE;                           // B_*U_

  k_conv4   <<<dim3(8, 2, 64), 256, 0, stream>>>(xs, ys, xe, ye, w1, b1, w2, b2,
                                                 w3, b3, w4, b4, outF, nat, tr);
  k_affinity<<<dim3(8, 8, 32), 256, 0, stream>>>(tr, Sx, Sy);
  k_softmax <<<dim3(B_ * P_), 256, 0, stream>>>(Sx, Sy);
  k_xdif    <<<dim3(8, 2, 16), 256, 0, stream>>>(w5, b5, nat, xdifF, xdifN);
  k_refine  <<<dim3(8, 2, 16), 256, 0, stream>>>(xdifN, Sx, xdifF, outF);
  k_pool    <<<dim3(B_ * U_), 256, 0, stream>>>(outF, pooled);
  k_mlp     <<<dim3(B_), 128, 0, stream>>>(pooled, lw1, lb1, lw2, lb2, outF);
}
